// GINConvNet_1245540516329
// MI455X (gfx1250) — matrix operand order repaired and checked by reading, not compiled, not run on hardware
//
#include <hip/hip_runtime.h>
#include <hip/hip_bf16.h>

// ---------------------------------------------------------------------------
// GINConvNet pipeline for gfx1250 (MI455X), bf16 WMMA everywhere GEMM-shaped.
// ---------------------------------------------------------------------------

typedef __attribute__((ext_vector_type(16))) __bf16 v16bf;
typedef __attribute__((ext_vector_type(16))) float  v16f;
typedef __attribute__((ext_vector_type(8)))  float  v8f;

__device__ __forceinline__ unsigned short f2bfu(float f) {
  unsigned int u = __float_as_uint(f);
  // round-to-nearest-even fp32 -> bf16 (used only in cold pack kernels)
  return (unsigned short)((u + 0x7FFFu + ((u >> 16) & 1u)) >> 16);
}

// Build an A fragment (16 bf16) from two 8-float contiguous runs.
__device__ __forceinline__ v16bf load_a_frag(const float* arp, int k0) {
  const float4* p0 = reinterpret_cast<const float4*>(arp + k0);
  const float4* p1 = reinterpret_cast<const float4*>(arp + k0 + 16);
  float4 x0 = p0[0], x1 = p0[1];
  float4 y0 = p1[0], y1 = p1[1];
  v16f av;
  av[0]  = x0.x; av[1]  = x0.y; av[2]  = x0.z; av[3]  = x0.w;
  av[4]  = x1.x; av[5]  = x1.y; av[6]  = x1.z; av[7]  = x1.w;
  av[8]  = y0.x; av[9]  = y0.y; av[10] = y0.z; av[11] = y0.w;
  av[12] = y1.x; av[13] = y1.y; av[14] = y1.z; av[15] = y1.w;
  return __builtin_convertvector(av, v16bf);  // packed v_cvt_pk_bf16_f32
}

// ---------------------------------------------------------------------------
// Pack a row-major fp32 matrix B[Ksrc x Nsrc] (ldb) into WMMA bf16 B-fragments.
// Layout per 16x16x32 step: lane = g*16+n holds B[ks*32 + (j<8 ? 8*g+j : 16+8*g+j-8), n], j=0..15,
// stored contiguously: dst[((ntile*ksteps + ks)*32 + lane)*16 + j].
// ---------------------------------------------------------------------------
__global__ void pack_b_kernel(const float* __restrict__ B, int ldb, int Ksrc,
                              int Nsrc, int ksteps, int ntiles,
                              unsigned short* __restrict__ dst) {
  int idx = blockIdx.x * blockDim.x + threadIdx.x;
  int total = ntiles * ksteps * 32;
  if (idx >= total) return;
  int lane = idx & 31;
  int ks   = (idx >> 5) % ksteps;
  int nt   = (idx >> 5) / ksteps;
  int g = lane >> 4;
  int n = nt * 16 + (lane & 15);
  unsigned short* d = dst + (long)idx * 16;
#pragma unroll
  for (int j = 0; j < 16; ++j) {
    int k = ks * 32 + (j < 8 ? g * 8 + j : 16 + g * 8 + (j - 8));
    float v = (k < Ksrc && n < Nsrc) ? B[(long)k * ldb + n] : 0.0f;
    d[j] = f2bfu(v);
  }
}

// ---------------------------------------------------------------------------
// Generic (optionally batched) WMMA GEMM: C = act(A * B + biasC + biasR).
// A: fp32 row-major [M x ksteps*32] (lda), zero-padded in K. M MUST be a
// multiple of 16 (true for all call sites). One wave computes a 16x32 tile.
// ---------------------------------------------------------------------------
__global__ __launch_bounds__(32)
void wmma_gemm_kernel(const float* __restrict__ A, int lda, long strideA,
                      const unsigned short* __restrict__ Bp, int ksteps,
                      const float* __restrict__ biasC,
                      const float* __restrict__ biasR,
                      float* __restrict__ C, int ldc, long strideC,
                      int Nlog, int relu) {
  const int lane = threadIdx.x;
  const int g = lane >> 4, l15 = lane & 15;
  const int m0 = blockIdx.x * 16;
  const int nt0 = blockIdx.y * 2;            // two consecutive 16-wide N tiles
  A += (long)blockIdx.z * strideA;
  C += (long)blockIdx.z * strideC;

  v8f acc0 = {}, acc1 = {};
  const float* arp = A + (long)(m0 + l15) * lda;
  const unsigned short* bp0 = Bp + ((long)nt0 * ksteps) * 512 + lane * 16;
  const unsigned short* bp1 = bp0 + (long)ksteps * 512;

  for (int ks = 0; ks < ksteps; ++ks) {
    // prefetch the B stream two k-steps ahead (global_prefetch_b8)
    if (ks + 2 < ksteps) {
      __builtin_prefetch(bp0 + (long)(ks + 2) * 512, 0, 1);
      __builtin_prefetch(bp1 + (long)(ks + 2) * 512, 0, 1);
    }
    v16bf bf0, bf1;
    __builtin_memcpy(&bf0, bp0 + (long)ks * 512, 32);
    __builtin_memcpy(&bf1, bp1 + (long)ks * 512, 32);
    v16bf af = load_a_frag(arp, ks * 32 + g * 8);
    acc0 = __builtin_amdgcn_wmma_f32_16x16x32_bf16(
        false, af, false, bf0, (short)0, acc0, false, false);
    acc1 = __builtin_amdgcn_wmma_f32_16x16x32_bf16(
        false, af, false, bf1, (short)0, acc1, false, false);
  }

  const int row0 = m0 + g * 8;
  const int col0 = nt0 * 16 + l15;
  if (col0 < Nlog) {
    const float bc = biasC ? biasC[col0] : 0.0f;
#pragma unroll
    for (int v = 0; v < 8; ++v) {
      const int row = row0 + v;
      float r = acc0[v] + bc + (biasR ? biasR[row] : 0.0f);
      if (relu) r = fmaxf(r, 0.0f);
      C[(long)row * ldc + col0] = r;
    }
  }
  const int col1 = col0 + 16;
  if (col1 < Nlog) {
    const float bc = biasC ? biasC[col1] : 0.0f;
#pragma unroll
    for (int v = 0; v < 8; ++v) {
      const int row = row0 + v;
      float r = acc1[v] + bc + (biasR ? biasR[row] : 0.0f);
      if (relu) r = fmaxf(r, 0.0f);
      C[(long)row * ldc + col1] = r;
    }
  }
}

// ---------------------------------------------------------------------------
// Node-MLP WMMA GEMM (Nlog == 32, ksteps <= 3): the whole bf16 B-pack
// (2*ksteps KB) is staged into LDS ONCE per workgroup with CDNA5 async
// tensor-path loads (global_load_async_to_lds_b128, ASYNCcnt), then 8 waves
// per block each compute one 16x32 M-tile reading B-fragments from LDS.
// ---------------------------------------------------------------------------
__global__ __launch_bounds__(256)
void wmma_gemm_lds_kernel(const float* __restrict__ A, int lda,
                          const unsigned short* __restrict__ Bp, int ksteps,
                          const float* __restrict__ biasC,
                          float* __restrict__ C, int ldc,
                          int mtiles, int relu) {
  __shared__ unsigned short ldsB[3 * 1024];  // max ksteps=3 -> 6144 B
  const int tid = threadIdx.x;

  // ---- async-stage B pack: ksteps*128 chunks of 16 bytes ----
  const int chunks = ksteps * 128;
  for (int c = tid; c < chunks; c += 256) {
    // low 32 bits of a generic LDS pointer == wave-relative LDS byte offset
    unsigned int loff = (unsigned int)(unsigned long long)(&ldsB[c * 8]);
    unsigned long long gaddr = (unsigned long long)(const void*)(Bp + c * 8);
    asm volatile("global_load_async_to_lds_b128 %0, %1, off"
                 :: "v"(loff), "v"(gaddr) : "memory");
  }
  asm volatile("s_wait_asynccnt 0x0" ::: "memory");
  __syncthreads();

  const int mtile = blockIdx.x * 8 + (tid >> 5);
  if (mtile >= mtiles) return;
  const int lane = tid & 31;
  const int g = lane >> 4, l15 = lane & 15;
  const int m0 = mtile * 16;

  v8f acc0 = {}, acc1 = {};
  const float* arp = A + (long)(m0 + l15) * lda;
  const unsigned short* lb0 = ldsB + lane * 16;                 // N-tile 0
  const unsigned short* lb1 = ldsB + (ksteps * 32 + lane) * 16; // N-tile 1

  for (int ks = 0; ks < ksteps; ++ks) {
    v16bf bf0, bf1;                     // ds_load_b128 pairs from LDS
    __builtin_memcpy(&bf0, lb0 + ks * 512, 32);
    __builtin_memcpy(&bf1, lb1 + ks * 512, 32);
    v16bf af = load_a_frag(arp, ks * 32 + g * 8);
    acc0 = __builtin_amdgcn_wmma_f32_16x16x32_bf16(
        false, af, false, bf0, (short)0, acc0, false, false);
    acc1 = __builtin_amdgcn_wmma_f32_16x16x32_bf16(
        false, af, false, bf1, (short)0, acc1, false, false);
  }

  const int row0 = m0 + g * 8;
  const float bc0 = biasC[l15];
  const float bc1 = biasC[16 + l15];
#pragma unroll
  for (int v = 0; v < 8; ++v) {
    const int row = row0 + v;
    float r0 = acc0[v] + bc0;
    float r1 = acc1[v] + bc1;
    if (relu) { r0 = fmaxf(r0, 0.0f); r1 = fmaxf(r1, 0.0f); }
    C[(long)row * ldc + l15]      = r0;
    C[(long)row * ldc + 16 + l15] = r1;
  }
}

// ---------------------------------------------------------------------------
// Small helpers
// ---------------------------------------------------------------------------
__global__ void zero_kernel(float* __restrict__ p, long n) {
  long i = blockIdx.x * (long)blockDim.x + threadIdx.x;
  for (; i < n; i += (long)gridDim.x * blockDim.x) p[i] = 0.0f;
}

// dst[N x lda] = src[N x F] zero-padded in columns (GIN self term: agg = x)
__global__ void copy_pad_kernel(const float* __restrict__ src, int F,
                                float* __restrict__ dst, int lda, long n) {
  long idx = blockIdx.x * (long)blockDim.x + threadIdx.x;
  long total = n * lda;
  if (idx >= total) return;
  long r = idx / lda;
  int  c = (int)(idx - r * lda);
  dst[idx] = (c < F) ? src[r * F + c] : 0.0f;
}

// agg[dst] += hin[src] over edges (segment_sum of neighbors)
__global__ void gin_scatter_kernel(const int* __restrict__ ei, int E,
                                   const float* __restrict__ hin, int ldin,
                                   float* __restrict__ agg, int lda, int F) {
  long idx = blockIdx.x * (long)blockDim.x + threadIdx.x;
  long total = (long)E * F;
  if (idx >= total) return;
  int e = (int)(idx / F);
  int f = (int)(idx - (long)e * F);
  int s = ei[e];        // edge_index[0, e]
  int d = ei[E + e];    // edge_index[1, e]
  atomicAdd(&agg[(long)d * lda + f], hin[(long)s * ldin + f]);
}

// column sums / sums-of-squares for BatchNorm (32 columns)
__global__ void bn_stats_kernel(const float* __restrict__ h, int n,
                                float* __restrict__ stats) {
  int col = threadIdx.x & 31;
  int rstart  = blockIdx.x * (blockDim.x >> 5) + (threadIdx.x >> 5);
  int rstride = gridDim.x * (blockDim.x >> 5);
  float s = 0.0f, s2 = 0.0f;
  for (int r = rstart; r < n; r += rstride) {
    float v = h[(long)r * 32 + col];
    s += v; s2 += v * v;
  }
  atomicAdd(&stats[col], s);
  atomicAdd(&stats[32 + col], s2);
}

__global__ void bn_finalize_kernel(float* __restrict__ stats,
                                   const float* __restrict__ gamma,
                                   const float* __restrict__ beta, float n) {
  int c = threadIdx.x;
  if (c < 32) {
    float mu  = stats[c] / n;
    float var = stats[32 + c] / n - mu * mu;       // biased variance
    float rstd = rsqrtf(var + 1e-5f);
    float sc = gamma[c] * rstd;
    stats[64 + c] = sc;                 // scale
    stats[96 + c] = beta[c] - mu * sc;  // shift
  }
}

__global__ void bn_apply_kernel(float* __restrict__ h,
                                const float* __restrict__ scale,
                                const float* __restrict__ shift, long total) {
  long idx = blockIdx.x * (long)blockDim.x + threadIdx.x;
  if (idx >= total) return;
  int c = (int)(idx & 31);
  h[idx] = fmaf(h[idx], scale[c], shift[c]);
}

// global_add_pool: g0[batch[n], c] += h[n, c]
__global__ void pool_kernel(const float* __restrict__ h,
                            const int* __restrict__ batch,
                            float* __restrict__ g0, int n) {
  long idx = blockIdx.x * (long)blockDim.x + threadIdx.x;
  if (idx >= (long)n * 32) return;
  int r = (int)(idx >> 5), c = (int)(idx & 31);
  atomicAdd(&g0[(long)batch[r] * 32 + c], h[idx]);
}

// W2[b,o,t*8+k] = sum_{i: target[b,i]==t} conv_w[o,i,k]  (26-symbol collapse)
// One block per b; thread owns one (o,k); LDS 32x208 accumulator.
__global__ void build_w2_kernel(const int* __restrict__ tgt,
                                const float* __restrict__ cw,
                                float* __restrict__ W2) {
  __shared__ float lds[32 * 208];
  const int b = blockIdx.x, tid = threadIdx.x;  // 256 threads
  const int o = tid >> 3, k = tid & 7;
  for (int i = tid; i < 32 * 208; i += 256) lds[i] = 0.0f;
  __syncthreads();
  const int* tb = tgt + (long)b * 1000;
  const float* cwo = cw + (long)o * 8000 + k;
  float* row = lds + o * 208 + k;
  for (int i = 0; i < 1000; ++i) {
    int t = tb[i];                 // uniform within block (broadcast)
    row[t * 8] += cwo[i * 8];      // unique (o,k) slot per thread -> no race
  }
  __syncthreads();
  float* outp = W2 + (long)b * 32 * 224;  // K padded 208 -> 224
  for (int i = tid; i < 32 * 224; i += 256) {
    int oo = i / 224, c = i - oo * 224;
    outp[i] = (c < 208) ? lds[oo * 208 + c] : 0.0f;
  }
}

// embB[t*8+k, h] = emb[t, h+k]   (shared conv B operand, 208 x 121)
__global__ void build_embB_kernel(const float* __restrict__ emb,
                                  float* __restrict__ eb) {
  int idx = blockIdx.x * blockDim.x + threadIdx.x;
  if (idx >= 208 * 121) return;
  int r = idx / 121, h = idx - r * 121;
  int t = r >> 3, k = r & 7;
  eb[idx] = emb[t * 128 + h + k];
}

// final projection [256,256] @ [256,1] + b
__global__ void out_kernel(const float* __restrict__ f2,
                           const float* __restrict__ ow,
                           const float* __restrict__ ob,
                           float* __restrict__ out) {
  int b = threadIdx.x;
  if (b < 256) {
    float s = ob[0];
    for (int j = 0; j < 256; ++j) s += f2[b * 256 + j] * ow[j];
    out[b] = s;
  }
}

// ---------------------------------------------------------------------------
// Host-side orchestration
// ---------------------------------------------------------------------------
extern "C" void kernel_launch(void* const* d_in, const int* in_sizes, int n_in,
                              void* d_out, int out_size, void* d_ws, size_t ws_size,
                              hipStream_t stream) {
  constexpr int N = 50000, E = 1000000, BGR = 256;
  constexpr int MT = 3125;                       // 50000/16 M-tiles
  constexpr unsigned MB = (MT + 7) / 8;          // blocks for 8-wave GEMM
  (void)in_sizes; (void)n_in; (void)out_size; (void)ws_size;

  const float* x        = (const float*)d_in[0];
  const int*   edge     = (const int*)  d_in[1];
  const int*   batch    = (const int*)  d_in[2];
  const int*   target   = (const int*)  d_in[3];
  const float* w1_0     = (const float*)d_in[4];
  const float* b1_0     = (const float*)d_in[5];
  const float* w2_0     = (const float*)d_in[6];
  const float* b2_0     = (const float*)d_in[7];
  const float* gamma0   = (const float*)d_in[8];
  const float* beta0    = (const float*)d_in[9];
  const float* w1s      = (const float*)d_in[10];
  const float* b1s      = (const float*)d_in[11];
  const float* w2s      = (const float*)d_in[12];
  const float* b2s      = (const float*)d_in[13];
  const float* gammas   = (const float*)d_in[14];
  const float* betas    = (const float*)d_in[15];
  const float* fc1_xd_w = (const float*)d_in[16];
  const float* fc1_xd_b = (const float*)d_in[17];
  const float* emb      = (const float*)d_in[18];
  const float* conv_w   = (const float*)d_in[19];
  const float* conv_b   = (const float*)d_in[20];
  const float* fc1_xt_w = (const float*)d_in[21];
  const float* fc1_xt_b = (const float*)d_in[22];
  const float* fc1_w    = (const float*)d_in[23];
  const float* fc1_b    = (const float*)d_in[24];
  const float* fc2_w    = (const float*)d_in[25];
  const float* fc2_b    = (const float*)d_in[26];
  const float* out_w    = (const float*)d_in[27];
  const float* out_b    = (const float*)d_in[28];
  float* out = (float*)d_out;

  // workspace carve-up (256B aligned)
  char* w = (char*)d_ws;
  size_t off = 0;
  auto alloc = [&](size_t bytes) -> void* {
    off = (off + 255) & ~(size_t)255;
    void* p = w + off;
    off += bytes;
    return p;
  };
  float* agg   = (float*)alloc((size_t)N * 96 * 4);
  float* h1    = (float*)alloc((size_t)N * 32 * 4);
  float* hcur  = (float*)alloc((size_t)N * 32 * 4);
  float* stats = (float*)alloc(128 * 4);
  float* g0    = (float*)alloc((size_t)BGR * 32 * 4);
  float* W2    = (float*)alloc((size_t)BGR * 32 * 224 * 4);
  float* embB  = (float*)alloc((size_t)208 * 121 * 4);
  float* cbuf  = (float*)alloc((size_t)BGR * 3872 * 4);
  float* xc    = (float*)alloc((size_t)BGR * 256 * 4);
  float* f1    = (float*)alloc((size_t)BGR * 1024 * 4);
  float* f2    = (float*)alloc((size_t)BGR * 256 * 4);
  unsigned short* pW1_0 = (unsigned short*)alloc((size_t)2 * 3 * 512 * 2);
  unsigned short* pW2_0 = (unsigned short*)alloc((size_t)2 * 1 * 512 * 2);
  unsigned short* pW1s  = (unsigned short*)alloc((size_t)4 * 1024 * 2);
  unsigned short* pW2s  = (unsigned short*)alloc((size_t)4 * 1024 * 2);
  unsigned short* pXD   = (unsigned short*)alloc((size_t)8 * 1 * 512 * 2);
  unsigned short* pEmb  = (unsigned short*)alloc((size_t)8 * 7 * 512 * 2);
  unsigned short* pXT   = (unsigned short*)alloc((size_t)8 * 121 * 512 * 2);
  unsigned short* pFC1  = (unsigned short*)alloc((size_t)64 * 8 * 512 * 2);
  unsigned short* pFC2  = (unsigned short*)alloc((size_t)16 * 32 * 512 * 2);

  auto packGrid = [](int ntiles, int ksteps) {
    return dim3((unsigned)((ntiles * ksteps * 32 + 255) / 256));
  };

  // ---- weight pre-packing to bf16 WMMA B-fragments ----
  pack_b_kernel<<<packGrid(2, 3), 256, 0, stream>>>(w1_0, 32, 78, 32, 3, 2, pW1_0);
  pack_b_kernel<<<packGrid(2, 1), 256, 0, stream>>>(w2_0, 32, 32, 32, 1, 2, pW2_0);
  for (int i = 0; i < 4; ++i) {
    pack_b_kernel<<<packGrid(2, 1), 256, 0, stream>>>(w1s + (size_t)i * 1024, 32, 32, 32, 1, 2, pW1s + (size_t)i * 1024);
    pack_b_kernel<<<packGrid(2, 1), 256, 0, stream>>>(w2s + (size_t)i * 1024, 32, 32, 32, 1, 2, pW2s + (size_t)i * 1024);
  }
  pack_b_kernel<<<packGrid(8, 1),   256, 0, stream>>>(fc1_xd_w, 128, 32, 128, 1, 8, pXD);
  pack_b_kernel<<<packGrid(8, 121), 256, 0, stream>>>(fc1_xt_w, 128, 3872, 128, 121, 8, pXT);
  pack_b_kernel<<<packGrid(64, 8),  256, 0, stream>>>(fc1_w, 1024, 256, 1024, 8, 64, pFC1);
  pack_b_kernel<<<packGrid(16, 32), 256, 0, stream>>>(fc2_w, 256, 1024, 256, 32, 16, pFC2);

  long tot;

  // ================= GIN layer 1 (78 -> 32 -> 32) =================
  tot = (long)N * 96;
  copy_pad_kernel<<<dim3((unsigned)((tot + 255) / 256)), 256, 0, stream>>>(x, 78, agg, 96, N);
  tot = (long)E * 78;
  gin_scatter_kernel<<<dim3((unsigned)((tot + 255) / 256)), 256, 0, stream>>>(edge, E, x, 78, agg, 96, 78);
  wmma_gemm_lds_kernel<<<MB, 256, 0, stream>>>(agg, 96, pW1_0, 3, b1_0, h1, 32, MT, 1);
  wmma_gemm_lds_kernel<<<MB, 256, 0, stream>>>(h1, 32, pW2_0, 1, b2_0, hcur, 32, MT, 1);
  zero_kernel<<<1, 128, 0, stream>>>(stats, 128);
  bn_stats_kernel<<<256, 256, 0, stream>>>(hcur, N, stats);
  bn_finalize_kernel<<<1, 32, 0, stream>>>(stats, gamma0, beta0, (float)N);
  bn_apply_kernel<<<dim3((unsigned)(((long)N * 32 + 255) / 256)), 256, 0, stream>>>(hcur, stats + 64, stats + 96, (long)N * 32);

  // ================= GIN layers 2..5 (32 -> 32 -> 32) =================
  for (int i = 0; i < 4; ++i) {
    tot = (long)N * 32;
    copy_pad_kernel<<<dim3((unsigned)((tot + 255) / 256)), 256, 0, stream>>>(hcur, 32, agg, 32, N);
    tot = (long)E * 32;
    gin_scatter_kernel<<<dim3((unsigned)((tot + 255) / 256)), 256, 0, stream>>>(edge, E, hcur, 32, agg, 32, 32);
    wmma_gemm_lds_kernel<<<MB, 256, 0, stream>>>(agg, 32, pW1s + (size_t)i * 1024, 1, b1s + (size_t)i * 32, h1, 32, MT, 1);
    wmma_gemm_lds_kernel<<<MB, 256, 0, stream>>>(h1, 32, pW2s + (size_t)i * 1024, 1, b2s + (size_t)i * 32, hcur, 32, MT, 1);
    zero_kernel<<<1, 128, 0, stream>>>(stats, 128);
    bn_stats_kernel<<<256, 256, 0, stream>>>(hcur, N, stats);
    bn_finalize_kernel<<<1, 32, 0, stream>>>(stats, gammas + (size_t)i * 32, betas + (size_t)i * 32, (float)N);
    bn_apply_kernel<<<dim3((unsigned)(((long)N * 32 + 255) / 256)), 256, 0, stream>>>(hcur, stats + 64, stats + 96, (long)N * 32);
  }

  // ================= pool + graph-side FC =================
  zero_kernel<<<32, 256, 0, stream>>>(g0, (long)BGR * 32);
  pool_kernel<<<dim3((unsigned)(((long)N * 32 + 255) / 256)), 256, 0, stream>>>(hcur, batch, g0, N);
  // g = relu(g0 @ fc1_xd_w + b) -> columns [0,128) of xc
  wmma_gemm_kernel<<<dim3(16, 4, 1), 32, 0, stream>>>(g0, 32, 0, pXD, 1, fc1_xd_b, nullptr, xc, 256, 0, 128, 1);

  // ================= conv branch (26-symbol collapse) =================
  build_w2_kernel<<<BGR, 256, 0, stream>>>(target, conv_w, W2);
  build_embB_kernel<<<dim3((208 * 121 + 255) / 256), 256, 0, stream>>>(emb, embB);
  pack_b_kernel<<<packGrid(8, 7), 256, 0, stream>>>(embB, 121, 208, 121, 7, 8, pEmb);
  // batched: c[b] = W2[b] @ embB + conv_b (per-row bias)
  wmma_gemm_kernel<<<dim3(2, 4, BGR), 32, 0, stream>>>(W2, 224, (long)32 * 224, pEmb, 7, nullptr, conv_b, cbuf, 121, 3872, 121, 0);
  // xt = c.reshape(256, 3872) @ fc1_xt_w + b -> columns [128,256) of xc
  wmma_gemm_kernel<<<dim3(16, 4, 1), 32, 0, stream>>>(cbuf, 3872, 0, pXT, 121, fc1_xt_b, nullptr, xc + 128, 256, 0, 128, 0);

  // ================= combined FC head =================
  wmma_gemm_kernel<<<dim3(16, 32, 1), 32, 0, stream>>>(xc, 256, 0, pFC1, 8, fc1_b, nullptr, f1, 1024, 0, 1024, 1);
  wmma_gemm_kernel<<<dim3(16, 8, 1), 32, 0, stream>>>(f1, 1024, 0, pFC2, 32, fc2_b, nullptr, f2, 256, 0, 256, 1);
  out_kernel<<<1, 256, 0, stream>>>(f2, out_w, out_b, out);
}
